// RWKV6_0_AttentionSubLayer_24532853195488
// MI455X (gfx1250) — matrix operand order repaired and checked by reading, not compiled, not run on hardware
//
#include <hip/hip_runtime.h>
#include <hip/hip_bf16.h>
#include <math.h>

// ---------------------------------------------------------------------------
// RWKV6 attention sublayer for MI455X (gfx1250, wave32, WMMA).
// - All large GEMMs: v_wmma_f32_16x16x32_bf16 (bf16 in, f32 acc).
// - Weights pre-transposed to padded [Npad][K] bf16 -> A and B tiles stage
//   identically as unconditional contiguous 32B async DMA transfers.
// - Double-buffered LDS, software-pipelined: steady-state loop body is
//   branch-free (stage next tile, s_wait_asynccnt 0x4, compute); last tile
//   peeled with s_wait_asynccnt 0x0.
// - Sequential WKV scan keeps the 64x64 per-(b,h) state in registers.
// ---------------------------------------------------------------------------

#define BB 4
#define TT 1024
#define CC 1024
#define HH 16
#define HEADSZ 64
#define TMD 32          // TIME_MIX_EXTRA_DIM
#define TDD 64          // W_MIX_EXTRA_DIM
#define MROWS (BB * TT) // 4096

#define USE_ASYNC_LDS 1

typedef __attribute__((ext_vector_type(16))) __bf16 v16bf;
typedef __attribute__((ext_vector_type(8)))  __bf16 v8bf;
typedef __attribute__((ext_vector_type(8)))  float  v8f;

// ---------------------------------------------------------------------------
// helpers
// ---------------------------------------------------------------------------
__device__ inline v16bf mk_frag(const __bf16* p0, const __bf16* p1) {
    v8bf lo = *(const v8bf*)p0;
    v8bf hi = *(const v8bf*)p1;
    v16bf r;
#pragma unroll
    for (int i = 0; i < 8; ++i) { r[i] = lo[i]; r[i + 8] = hi[i]; }
    return r;
}

__device__ inline v8f wmma_bf16(v16bf a, v16bf b, v8f c) {
    // D = A(16x32 bf16) x B(32x16 bf16) + C(16x16 f32)
    return __builtin_amdgcn_wmma_f32_16x16x32_bf16(
        /*neg_a=*/false, a, /*neg_b=*/false, b,
        /*c_mod=*/(short)0, c, /*reuse_a=*/false, /*reuse_b=*/false);
}

// 32 bytes global -> LDS via async DMA (2 x b128; INST_OFFSET advances both
// the global and the LDS address, so one address pair serves both beats).
__device__ inline void stage32B(const __bf16* g, __bf16* l) {
#if USE_ASYNC_LDS
    unsigned           lds = (unsigned)(uintptr_t)(void*)l;   // low 32b = LDS offset
    unsigned long long ga  = (unsigned long long)(uintptr_t)g;
    asm volatile("global_load_async_to_lds_b128 %0, %1, off\n\t"
                 "global_load_async_to_lds_b128 %0, %1, off offset:16"
                 :: "v"(lds), "v"(ga) : "memory");
#else
    *(v8bf*)l       = *(const v8bf*)g;
    *(v8bf*)(l + 8) = *(const v8bf*)(g + 8);
#endif
}

// wait until at most `4` / `0` async ops remain outstanding (per-wave, in-order)
__device__ inline void wait_async4() {
#if USE_ASYNC_LDS
    asm volatile("s_wait_asynccnt 0x4" ::: "memory");
#endif
}
__device__ inline void wait_async0() {
#if USE_ASYNC_LDS
    asm volatile("s_wait_asynccnt 0x0" ::: "memory");
#endif
}

enum { EPI_NONE = 0, EPI_TANH = 1, EPI_SILU = 2 };

template <int EPI>
__device__ inline float epilogue(float x) {
    if (EPI == EPI_TANH) return tanhf(x);
    if (EPI == EPI_SILU) return x / (1.0f + expf(-x));
    return x;
}

// ---------------------------------------------------------------------------
// tiled f32[K,N] -> bf16[N,K] transpose-convert (K,N multiples of 32 here)
// ---------------------------------------------------------------------------
__global__ __launch_bounds__(256)
void cvt_transpose_kernel(const float* __restrict__ src, __bf16* __restrict__ dst,
                          int K, int N) {
    __shared__ float tile[32][33];
    const int kb = blockIdx.y * 32;
    const int nb = blockIdx.x * 32;
#pragma unroll
    for (int i = threadIdx.y; i < 32; i += 8)
        tile[i][threadIdx.x] = src[(size_t)(kb + i) * N + nb + threadIdx.x];
    __syncthreads();
#pragma unroll
    for (int i = threadIdx.y; i < 32; i += 8)
        dst[(size_t)(nb + i) * K + kb + threadIdx.x] = (__bf16)tile[threadIdx.x][i];
}

// zero-fill bf16 (for weight row padding)
__global__ void zero_bf16_kernel(__bf16* __restrict__ p, int n) {
    int i = blockIdx.x * blockDim.x + threadIdx.x;
    if (i < n) p[i] = (__bf16)0.0f;
}

// ---------------------------------------------------------------------------
// token shift mix:  xm = bf16(x + (x_prev - x) * x_maa)
// ---------------------------------------------------------------------------
__global__ void mix1_kernel(const float* __restrict__ x,
                            const float* __restrict__ x_maa,
                            __bf16* __restrict__ xm) {
    int idx = blockIdx.x * blockDim.x + threadIdx.x;  // < 4M
    int c = idx & (CC - 1);
    int row = idx >> 10;
    int t = row & (TT - 1);
    float xv = x[idx];
    float sx = (t == 0 ? 0.0f : x[idx - CC]) - xv;
    xm[idx] = (__bf16)(xv + sx * x_maa[c]);
}

// ---------------------------------------------------------------------------
// bf16 WMMA GEMM:  C[M,N] = epi( A[M,K] x BT[Npad,K]^T ),  Npad = ceil128(N)
// Block: 256 threads (8 waves), tile BM=128 x BN=128, K-step 32,
// double-buffered LDS with async-to-LDS DMA overlap, last tile peeled.
// Wave grid 4x2; each wave owns 32(M) x 64(N): 2x4 WMMA accumulators.
// Fragment layouts per ISA 7.12.2:
//   A 16x32: lane row=l&15; elems 0..7 K=half*8.., 8..15 K=16+half*8..
//   B 32x16: lane col=l&15; elems 0..7 K=half*8.., 8..15 K=16+half*8.. (same as A)
// ---------------------------------------------------------------------------
constexpr int GBM = 128, GBN = 128, GKT = 32;
constexpr int GS = 40;  // padded LDS row stride (elements); 80B rows, 16B aligned

__device__ inline void gemm_compute_step(const __bf16* __restrict__ Asp,
                                         const __bf16* __restrict__ Bsp,
                                         int wm, int wn, int l15, int half,
                                         v8f acc[2][4]) {
    v16bf afrag[2];
#pragma unroll
    for (int i = 0; i < 2; ++i) {
        const __bf16* ar = Asp + (wm * 32 + i * 16 + l15) * GS;
        afrag[i] = mk_frag(ar + half * 8, ar + 16 + half * 8);
    }
    v16bf bfrag[4];
#pragma unroll
    for (int j = 0; j < 4; ++j) {
        const __bf16* br = Bsp + (wn * 64 + j * 16 + l15) * GS;
        bfrag[j] = mk_frag(br + half * 8, br + 16 + half * 8);
    }
#pragma unroll
    for (int i = 0; i < 2; ++i)
#pragma unroll
        for (int j = 0; j < 4; ++j)
            acc[i][j] = wmma_bf16(afrag[i], bfrag[j], acc[i][j]);
}

template <int EPI>
__global__ __launch_bounds__(256)
void gemm_bf16_wmma(const __bf16* __restrict__ A,
                    const __bf16* __restrict__ BT,
                    float* __restrict__ Cm,
                    int M, int N, int K) {
    __shared__ __attribute__((aligned(16))) __bf16 As[2][GBM * GS];
    __shared__ __attribute__((aligned(16))) __bf16 Bs[2][GBN * GS];

    const int tid  = threadIdx.x;
    const int wave = tid >> 5;
    const int lane = tid & 31;
    const int Mb = blockIdx.y * GBM;
    const int Nb = blockIdx.x * GBN;
    const int wm = wave >> 1;   // 0..3
    const int wn = wave & 1;    // 0..1

    v8f acc[2][4];
#pragma unroll
    for (int i = 0; i < 2; ++i)
#pragma unroll
        for (int j = 0; j < 4; ++j)
#pragma unroll
            for (int r = 0; r < 8; ++r) acc[i][j][r] = 0.0f;

    // staging assignment: 256 threads cover 128 rows x (2 chunks of 16 elems)
    const int sr = tid >> 1;            // 0..127
    const int sc = (tid & 1) * 16;      // 0 or 16
    const __bf16* ga = A  + (size_t)(Mb + sr) * K + sc;
    const __bf16* gb = BT + (size_t)(Nb + sr) * K + sc;
    const int lofs = sr * GS + sc;

    const int half = lane >> 4;    // 0 or 1
    const int l15  = lane & 15;

    const int nk = K / GKT;        // >= 2 for all our GEMMs (K = 1024)
    // prologue: stage tile 0 into buffer 0 (4 async ops per thread)
    stage32B(ga, &As[0][lofs]);
    stage32B(gb, &Bs[0][lofs]);

    // steady state: branch-free body, last iteration peeled
    for (int kt = 0; kt < nk - 1; ++kt) {
        const int p = kt & 1;
        stage32B(ga + (kt + 1) * GKT, &As[p ^ 1][lofs]);
        stage32B(gb + (kt + 1) * GKT, &Bs[p ^ 1][lofs]);
        wait_async4();   // previous tile's 4 ops done; next 4 still in flight
        __syncthreads();
        gemm_compute_step(&As[p][0], &Bs[p][0], wm, wn, l15, half, acc);
        __syncthreads();   // all waves done with buffer p before it is restaged
    }
    {   // final tile
        const int p = (nk - 1) & 1;
        wait_async0();
        __syncthreads();
        gemm_compute_step(&As[p][0], &Bs[p][0], wm, wn, l15, half, acc);
    }

    // --- store: C/D layout: lane n = lane&15; VGPR r -> M = r + 8*(lane>>4)
#pragma unroll
    for (int i = 0; i < 2; ++i)
#pragma unroll
        for (int j = 0; j < 4; ++j) {
            const int n = Nb + wn * 64 + j * 16 + l15;
            if (n < N) {
#pragma unroll
                for (int r = 0; r < 8; ++r) {
                    const int m = Mb + wm * 32 + i * 16 + half * 8 + r;
                    Cm[(size_t)m * N + n] = epilogue<EPI>(acc[i][j][r]);
                }
            }
        }
}

// ---------------------------------------------------------------------------
// mixing-delta expansion + 5 mixed activations (bf16 for WMMA consumers)
// z[row,160] (tanh applied in GEMM epilogue), tm_w2[5,32,C]
// ---------------------------------------------------------------------------
__global__ __launch_bounds__(256)
void mix2_kernel(const float* __restrict__ x,
                 const float* __restrict__ z,
                 const float* __restrict__ tm_w2,
                 const float* __restrict__ w_maa,
                 const float* __restrict__ k_maa,
                 const float* __restrict__ v_maa,
                 const float* __restrict__ r_maa,
                 const float* __restrict__ g_maa,
                 __bf16* __restrict__ wxb, __bf16* __restrict__ kxb,
                 __bf16* __restrict__ vxb, __bf16* __restrict__ rxb,
                 __bf16* __restrict__ gxb) {
    const int row = blockIdx.y;
    const int c = blockIdx.x * 256 + threadIdx.x;
    __shared__ float zs[5 * TMD];
    if (threadIdx.x < 5 * TMD) zs[threadIdx.x] = z[row * (5 * TMD) + threadIdx.x];
    __syncthreads();

    const int t = row & (TT - 1);
    const size_t idx = (size_t)row * CC + c;
    const float xv = x[idx];
    const float sx = (t == 0 ? 0.0f : x[idx - CC]) - xv;

    float m[5];
#pragma unroll
    for (int f = 0; f < 5; ++f) {
        float accv = 0.0f;
#pragma unroll
        for (int d = 0; d < TMD; ++d)
            accv += zs[f * TMD + d] * tm_w2[(size_t)(f * TMD + d) * CC + c];
        m[f] = accv;
    }
    wxb[idx] = (__bf16)(xv + sx * (w_maa[c] + m[0]));
    kxb[idx] = (__bf16)(xv + sx * (k_maa[c] + m[1]));
    vxb[idx] = (__bf16)(xv + sx * (v_maa[c] + m[2]));
    rxb[idx] = (__bf16)(xv + sx * (r_maa[c] + m[3]));
    gxb[idx] = (__bf16)(xv + sx * (g_maa[c] + m[4]));
}

// ---------------------------------------------------------------------------
// decay: w = exp(-exp(time_decay + tanh(wx@td_w1) @ td_w2)); td_w2[64,C]
// ---------------------------------------------------------------------------
__global__ __launch_bounds__(256)
void decay_kernel(const float* __restrict__ zw,
                  const float* __restrict__ td_w2,
                  const float* __restrict__ time_decay,
                  float* __restrict__ wout) {
    const int row = blockIdx.y;
    const int c = blockIdx.x * 256 + threadIdx.x;
    __shared__ float zs[TDD];
    if (threadIdx.x < TDD) zs[threadIdx.x] = zw[row * TDD + threadIdx.x];
    __syncthreads();

    float accv = time_decay[c];
#pragma unroll
    for (int d = 0; d < TDD; ++d)
        accv += zs[d] * td_w2[(size_t)d * CC + c];
    wout[(size_t)row * CC + c] = expf(-expf(accv));
}

// ---------------------------------------------------------------------------
// WKV sequential scan. One block per (b,h); 64 threads, thread = V column.
// ---------------------------------------------------------------------------
__global__ __launch_bounds__(64)
void wkv_scan_kernel(const float* __restrict__ r,
                     const float* __restrict__ k,
                     const float* __restrict__ v,
                     const float* __restrict__ w,
                     const float* __restrict__ u,
                     float* __restrict__ y) {
    const int bh = blockIdx.x;
    const int b = bh / HH;
    const int h = bh % HH;
    const int tid = threadIdx.x;  // 0..63

    __shared__ float rS[HEADSZ], kS[HEADSZ], wS[HEADSZ], uS[HEADSZ];
    uS[tid] = u[h * HEADSZ + tid];

    float s[HEADSZ];
#pragma unroll
    for (int i = 0; i < HEADSZ; ++i) s[i] = 0.0f;

    for (int t = 0; t < TT; ++t) {
        const size_t base = ((size_t)(b * TT + t)) * CC + h * HEADSZ;
        __syncthreads();                 // previous step done with LDS
        rS[tid] = r[base + tid];
        kS[tid] = k[base + tid];
        wS[tid] = w[base + tid];
        const float vv = v[base + tid];
        __syncthreads();

        float yv = 0.0f;
#pragma unroll
        for (int kk = 0; kk < HEADSZ; ++kk) {
            const float kv = kS[kk] * vv;
            yv += rS[kk] * (s[kk] + uS[kk] * kv);
            s[kk] = wS[kk] * s[kk] + kv;
        }
        y[base + tid] = yv * 0.125f;     // / head_size_divisor
    }
}

// ---------------------------------------------------------------------------
// per-head GroupNorm + affine + gate multiply -> bf16 activation for Wo GEMM
// ---------------------------------------------------------------------------
__global__ __launch_bounds__(256)
void gn_gate_kernel(const float* __restrict__ y,
                    const float* __restrict__ g,     // silu already applied
                    const float* __restrict__ ln_w,
                    const float* __restrict__ ln_b,
                    __bf16* __restrict__ abf) {
    const int row = blockIdx.x;
    const int tid = threadIdx.x;
    __shared__ float ys[CC];
    __shared__ float muS[HH], rsS[HH];

    for (int c = tid; c < CC; c += 256) ys[c] = y[(size_t)row * CC + c];
    __syncthreads();

    if (tid < HH) {
        float sum = 0.0f, sq = 0.0f;
#pragma unroll
        for (int i = 0; i < HEADSZ; ++i) {
            float xv = ys[tid * HEADSZ + i];
            sum += xv; sq += xv * xv;
        }
        float mu = sum * (1.0f / HEADSZ);
        float var = sq * (1.0f / HEADSZ) - mu * mu;
        muS[tid] = mu;
        rsS[tid] = rsqrtf(var + 1e-5f);
    }
    __syncthreads();

    for (int c = tid; c < CC; c += 256) {
        int h = c >> 6;
        float val = (ys[c] - muS[h]) * rsS[h] * ln_w[c] + ln_b[c];
        abf[(size_t)row * CC + c] = (__bf16)(val * g[(size_t)row * CC + c]);
    }
}

// ---------------------------------------------------------------------------
// host launch
// ---------------------------------------------------------------------------
extern "C" void kernel_launch(void* const* d_in, const int* in_sizes, int n_in,
                              void* d_out, int out_size, void* d_ws, size_t ws_size,
                              hipStream_t stream) {
    const float* x      = (const float*)d_in[0];
    const float* x_maa  = (const float*)d_in[1];
    const float* w_maa  = (const float*)d_in[2];
    const float* k_maa  = (const float*)d_in[3];
    const float* v_maa  = (const float*)d_in[4];
    const float* r_maa  = (const float*)d_in[5];
    const float* g_maa  = (const float*)d_in[6];
    const float* tm_w1  = (const float*)d_in[7];
    const float* tm_w2  = (const float*)d_in[8];
    const float* td_w1  = (const float*)d_in[9];
    const float* td_w2  = (const float*)d_in[10];
    const float* tdecay = (const float*)d_in[11];
    const float* tfirst = (const float*)d_in[12];
    const float* Wr     = (const float*)d_in[13];
    const float* Wk     = (const float*)d_in[14];
    const float* Wv     = (const float*)d_in[15];
    const float* Wg     = (const float*)d_in[16];
    const float* Wo     = (const float*)d_in[17];
    const float* ln_w   = (const float*)d_in[18];
    const float* ln_b   = (const float*)d_in[19];
    float* out = (float*)d_out;

    // --------- workspace carve-out (aligned 256B) ---------------------------
    size_t off = 0;
    auto carve = [&](size_t bytes) -> void* {
        void* p = (char*)d_ws + off;
        off = (off + bytes + 255) & ~(size_t)255;
        return p;
    };
    const size_t NMAT = (size_t)CC * CC;           // 1M
    const size_t NACT = (size_t)MROWS * CC;        // 4M
    const int NTM_PAD = 256;                       // ceil128(160)
    const int NTD_PAD = 128;                       // ceil128(64)

    __bf16* wbr  = (__bf16*)carve(NMAT * 2);                 // [1024][1024]
    __bf16* wbk  = (__bf16*)carve(NMAT * 2);
    __bf16* wbv  = (__bf16*)carve(NMAT * 2);
    __bf16* wbg  = (__bf16*)carve(NMAT * 2);
    __bf16* wbo  = (__bf16*)carve(NMAT * 2);
    __bf16* wbtm = (__bf16*)carve((size_t)NTM_PAD * CC * 2); // [256][1024] padded
    __bf16* wbtd = (__bf16*)carve((size_t)NTD_PAD * CC * 2); // [128][1024] padded
    __bf16* xm   = (__bf16*)carve(NACT * 2);
    float*  zbuf = (float*) carve((size_t)MROWS * 5 * TMD * 4);
    __bf16* wxb  = (__bf16*)carve(NACT * 2);
    __bf16* kxb  = (__bf16*)carve(NACT * 2);
    __bf16* vxb  = (__bf16*)carve(NACT * 2);
    __bf16* rxb  = (__bf16*)carve(NACT * 2);
    __bf16* gxb  = (__bf16*)carve(NACT * 2);
    float*  zw   = (float*) carve((size_t)MROWS * TDD * 4);
    float*  wdec = (float*) carve(NACT * 4);
    float*  rf   = (float*) carve(NACT * 4);
    float*  kf   = (float*) carve(NACT * 4);
    float*  vf   = (float*) carve(NACT * 4);
    float*  gf   = (float*) carve(NACT * 4);
    float*  yf   = (float*) carve(NACT * 4);
    __bf16* abf  = (__bf16*)carve(NACT * 2);
    (void)ws_size; (void)n_in; (void)in_sizes; (void)out_size;

    // --------- 1) weight transpose-converts to bf16 [N][K]; zero row pad ----
    auto cvtT = [&](const float* s, __bf16* d, int K, int N) {
        cvt_transpose_kernel<<<dim3(N / 32, K / 32), dim3(32, 8), 0, stream>>>(s, d, K, N);
    };
    cvtT(Wr, wbr, CC, CC); cvtT(Wk, wbk, CC, CC); cvtT(Wv, wbv, CC, CC);
    cvtT(Wg, wbg, CC, CC); cvtT(Wo, wbo, CC, CC);
    cvtT(tm_w1, wbtm, CC, 5 * TMD);
    cvtT(td_w1, wbtd, CC, TDD);
    {   // zero padding rows [N..Npad) so B staging needs no guards
        int ntm = (NTM_PAD - 5 * TMD) * CC;   // 96*1024
        int ntd = (NTD_PAD - TDD) * CC;       // 64*1024
        zero_bf16_kernel<<<(ntm + 255) / 256, 256, 0, stream>>>(wbtm + (size_t)(5 * TMD) * CC, ntm);
        zero_bf16_kernel<<<(ntd + 255) / 256, 256, 0, stream>>>(wbtd + (size_t)TDD * CC, ntd);
    }

    // --------- 2) token-shift mix for LoRA input ----------------------------
    mix1_kernel<<<(unsigned)(NACT / 256), 256, 0, stream>>>(x, x_maa, xm);

    // --------- 3) z = tanh(xm @ tm_w1)  [4096 x 160] ------------------------
    gemm_bf16_wmma<EPI_TANH><<<dim3(2, MROWS / 128), 256, 0, stream>>>(
        xm, wbtm, zbuf, MROWS, 5 * TMD, CC);

    // --------- 4) five mixed activations ------------------------------------
    mix2_kernel<<<dim3(CC / 256, MROWS), 256, 0, stream>>>(
        x, zbuf, tm_w2, w_maa, k_maa, v_maa, r_maa, g_maa,
        wxb, kxb, vxb, rxb, gxb);

    // --------- 5) zw = tanh(wx @ td_w1)  [4096 x 64] ------------------------
    gemm_bf16_wmma<EPI_TANH><<<dim3(1, MROWS / 128), 256, 0, stream>>>(
        wxb, wbtd, zw, MROWS, TDD, CC);

    // --------- 6) w = exp(-exp(time_decay + zw @ td_w2)) --------------------
    decay_kernel<<<dim3(CC / 256, MROWS), 256, 0, stream>>>(
        zw, td_w2, tdecay, wdec);

    // --------- 7) projections r,k,v and gated g -----------------------------
    gemm_bf16_wmma<EPI_NONE><<<dim3(CC / 128, MROWS / 128), 256, 0, stream>>>(
        rxb, wbr, rf, MROWS, CC, CC);
    gemm_bf16_wmma<EPI_NONE><<<dim3(CC / 128, MROWS / 128), 256, 0, stream>>>(
        kxb, wbk, kf, MROWS, CC, CC);
    gemm_bf16_wmma<EPI_NONE><<<dim3(CC / 128, MROWS / 128), 256, 0, stream>>>(
        vxb, wbv, vf, MROWS, CC, CC);
    gemm_bf16_wmma<EPI_SILU><<<dim3(CC / 128, MROWS / 128), 256, 0, stream>>>(
        gxb, wbg, gf, MROWS, CC, CC);

    // --------- 8) sequential WKV scan ---------------------------------------
    wkv_scan_kernel<<<BB * HH, HEADSZ, 0, stream>>>(rf, kf, vf, wdec, tfirst, yf);

    // --------- 9) per-head GroupNorm + gate ---------------------------------
    gn_gate_kernel<<<MROWS, 256, 0, stream>>>(yf, gf, ln_w, ln_b, abf);

    // --------- 10) output projection ----------------------------------------
    gemm_bf16_wmma<EPI_NONE><<<dim3(CC / 128, MROWS / 128), 256, 0, stream>>>(
        abf, wbo, out, MROWS, CC, CC);
}
